// OW_CAM_56384330662297
// MI455X (gfx1250) — matrix operand order repaired and checked by reading, not compiled, not run on hardware
//
#include <hip/hip_runtime.h>
#include <hip/hip_bf16.h>
#include <stdint.h>

// ---------------------------------------------------------------------------
// Types for CDNA5 WMMA
// ---------------------------------------------------------------------------
typedef __attribute__((ext_vector_type(16))) __bf16 v16bf;
typedef __attribute__((ext_vector_type(8)))  float  v8f;

union BF16x16 { v16bf v; uint4 q[2]; };

#define WMMA_BF16(A, B, C) \
    __builtin_amdgcn_wmma_f32_16x16x32_bf16(false, (A), false, (B), (short)0, (C), false, false)

// async global->LDS copy path (CDNA5 GLOBAL_LOAD_ASYNC_TO_LDS_B128), if the
// toolchain exposes the builtin; falls back to VGPR-staged copies otherwise.
#if defined(__has_builtin)
#  if __has_builtin(__builtin_amdgcn_global_load_async_to_lds_b128)
#    define OW_ASYNC_LDS 1
#  endif
#endif
#ifndef OW_ASYNC_LDS
#  define OW_ASYNC_LDS 0
#endif

// parameter type revealed by clang diagnostic: AS1/AS3 pointers to
// int __attribute__((vector_size(16)))
typedef int owint4 __attribute__((vector_size(16)));
typedef __attribute__((address_space(1))) owint4 ow_g4;
typedef __attribute__((address_space(3))) owint4 ow_l4;

__device__ __forceinline__ unsigned short f2bf_u(float f) {
    union { float f; uint32_t u; } c; c.f = f;
    uint32_t u = c.u + 0x7fffu + ((c.u >> 16) & 1u);   // round-to-nearest-even
    return (unsigned short)(u >> 16);
}
__device__ __forceinline__ float bf2f_u(unsigned short s) {
    union { uint32_t u; float f; } c; c.u = ((uint32_t)s) << 16;
    return c.f;
}

#define NTOK   16384
#define CDIM   192
#define NHEAD  6
#define HD     32
#define NQ     256       // 16*16 queries per window
#define NK     576       // 24*24 keys per window
#define NWIN   64
#define MH_DIM 384
#define NEXP   4
#define SCALE_Q 0.17677669529663689f   // 32^-0.5

// ---------------------------------------------------------------------------
// LayerNorm (f32 in) -> bf16 out.  One block (192 threads) per row.
// ---------------------------------------------------------------------------
__global__ void ow_ln_kernel(const float* __restrict__ x,
                             const float* __restrict__ g,
                             const float* __restrict__ b,
                             unsigned short* __restrict__ out) {
    __shared__ float s1[192];
    __shared__ float s2[192];
    int row = blockIdx.x;
    int t = threadIdx.x;
    float v = x[(size_t)row * CDIM + t];
    s1[t] = v; s2[t] = v * v;
    __syncthreads();
    if (t < 64) { s1[t] += s1[t + 64] + s1[t + 128]; s2[t] += s2[t + 64] + s2[t + 128]; }
    __syncthreads();
    for (int s = 32; s > 0; s >>= 1) {
        if (t < s) { s1[t] += s1[t + s]; s2[t] += s2[t + s]; }
        __syncthreads();
    }
    float mu  = s1[0] * (1.0f / 192.0f);
    float var = s2[0] * (1.0f / 192.0f) - mu * mu;
    float inv = rsqrtf(var + 1e-5f);
    out[(size_t)row * CDIM + t] = f2bf_u((v - mu) * inv * g[t] + b[t]);
}

// ---------------------------------------------------------------------------
// Generic WMMA GEMM:  D = epilogue(A[MxK](bf16) @ B[KxN](f32 weights) + bias)
// block tile 256x64, 256 threads = 8 waves arranged 4(M) x 2(N),
// per-wave tile 64x32 -> 8 WMMAs per K-chunk (4 A-tiles x 2 B-tiles).
// modes: 0=split QV (bf16 x2, scale first 192 cols)
//        1=residual add, f32 out
//        2=GELU -> bf16
//        3=plain -> bf16
// ---------------------------------------------------------------------------
__global__ __launch_bounds__(256)
void ow_gemm_wmma(const unsigned short* __restrict__ A,
                  const float* __restrict__ B,
                  const float* __restrict__ bias,
                  int M, int N, int K,
                  int mode, float scale0,
                  unsigned short* __restrict__ ob0,
                  unsigned short* __restrict__ ob1,
                  const float* __restrict__ res,
                  float* __restrict__ of) {
    __shared__ __align__(16) unsigned short As[256 * 32];  // [row][k]
    __shared__ __align__(16) unsigned short Bt[64 * 32];   // [n][k] (transposed)

    const int bm = blockIdx.x, bn = blockIdx.y;
    const int tid = threadIdx.x;
    const int lane = tid & 31, wave = tid >> 5;
    const int wm = wave >> 1, wn = wave & 1;     // 4 x 2 wave grid
    const int half = lane >> 4, l16 = lane & 15;

    v8f acc[4][2] = {};

    for (int k0 = 0; k0 < K; k0 += 32) {
        // --- stage A tile 256x32 bf16 (64B per thread: one full row slice) ---
#if OW_ASYNC_LDS
        {
            const unsigned short* gsrc = A + (size_t)(bm * 256 + tid) * K + k0;
            unsigned short* ldst = As + tid * 32;
            __builtin_amdgcn_global_load_async_to_lds_b128((ow_g4*)gsrc, (ow_l4*)ldst, 0,  0);
            __builtin_amdgcn_global_load_async_to_lds_b128((ow_g4*)gsrc, (ow_l4*)ldst, 16, 0);
            __builtin_amdgcn_global_load_async_to_lds_b128((ow_g4*)gsrc, (ow_l4*)ldst, 32, 0);
            __builtin_amdgcn_global_load_async_to_lds_b128((ow_g4*)gsrc, (ow_l4*)ldst, 48, 0);
            if (k0 + 32 < K)   // prefetch next A chunk (global_prefetch_b8)
                __builtin_prefetch((const char*)gsrc + 64, 0, 1);
        }
#else
        {
            const uint4* src = (const uint4*)(A + (size_t)(bm * 256 + tid) * K + k0);
            uint4 t0 = src[0], t1 = src[1], t2 = src[2], t3 = src[3];
            uint4* dst = (uint4*)(As + tid * 32);
            dst[0] = t0; dst[1] = t1; dst[2] = t2; dst[3] = t3;
            if (k0 + 32 < K)   // prefetch next A chunk (global_prefetch_b8)
                __builtin_prefetch((const char*)(src + 4), 0, 1);
        }
#endif
        // --- load B tile 32x64 f32, convert+transpose into Bt[n][k] ---
        {
            int r = tid >> 3, cs = (tid & 7) * 8;
            const float* src = B + (size_t)(k0 + r) * N + bn * 64 + cs;
#pragma unroll
            for (int j = 0; j < 8; ++j)
                Bt[(cs + j) * 32 + r] = f2bf_u(src[j]);
            if (k0 + 32 < K)
                __builtin_prefetch((const char*)(src + (size_t)32 * N), 0, 1);
        }
#if OW_ASYNC_LDS
        // async LDS writes are tracked by ASYNCcnt, not DScnt: drain before the
        // barrier so all waves see the tile after the barrier completes.
        asm volatile("s_wait_asynccnt 0" ::: "memory");
#endif
        __syncthreads();

        BF16x16 a[4], bb[2];
#pragma unroll
        for (int i = 0; i < 4; ++i) {
            int m = wm * 64 + i * 16 + l16;
            // A layout: K = half*8 + {0..7} and half*8 + 16 + {0..7}
            a[i].q[0] = *(const uint4*)(As + m * 32 + half * 8);
            a[i].q[1] = *(const uint4*)(As + m * 32 + half * 8 + 16);
        }
#pragma unroll
        for (int j = 0; j < 2; ++j) {
            int n = wn * 32 + j * 16 + l16;
            // B layout: K = half*8 + {0..7} and half*8 + 16 + {0..7}
            bb[j].q[0] = *(const uint4*)(Bt + n * 32 + half * 8);
            bb[j].q[1] = *(const uint4*)(Bt + n * 32 + half * 8 + 16);
        }
#pragma unroll
        for (int i = 0; i < 4; ++i)
#pragma unroll
            for (int j = 0; j < 2; ++j)
                acc[i][j] = WMMA_BF16(a[i].v, bb[j].v, acc[i][j]);
        __syncthreads();
    }

    // epilogue: element (r) -> row m_local = r + 8*half, col = l16
#pragma unroll
    for (int i = 0; i < 4; ++i) {
#pragma unroll
        for (int j = 0; j < 2; ++j) {
            int n = bn * 64 + wn * 32 + j * 16 + l16;
            float bv = bias[n];
#pragma unroll
            for (int r = 0; r < 8; ++r) {
                int m = bm * 256 + wm * 64 + i * 16 + r + half * 8;
                float val = acc[i][j][r] + bv;
                if (mode == 0) {
                    if (n < CDIM) ob0[(size_t)m * CDIM + n] = f2bf_u(val * scale0);
                    else          ob1[(size_t)m * CDIM + (n - CDIM)] = f2bf_u(val);
                } else if (mode == 1) {
                    of[(size_t)m * N + n] = val + res[(size_t)m * N + n];
                } else if (mode == 2) {
                    float gv = 0.5f * val * (1.0f + erff(val * 0.70710678118654752f));
                    ob0[(size_t)m * N + n] = f2bf_u(gv);
                } else {
                    ob0[(size_t)m * N + n] = f2bf_u(val);
                }
            }
        }
    }
}

// ---------------------------------------------------------------------------
// Precompute attention bias table: bias[h][q][k] = rpb[rpi[q*576+k]*6 + h]
// ---------------------------------------------------------------------------
__global__ void ow_bias_pre(const int* __restrict__ rpi,
                            const float* __restrict__ rpb,
                            float* __restrict__ bias) {
    int i = blockIdx.x * blockDim.x + threadIdx.x;
    const int total = NHEAD * NQ * NK;
    if (i >= total) return;
    int h  = i / (NQ * NK);
    int qk = i - h * (NQ * NK);
    bias[i] = rpb[rpi[qk] * NHEAD + h];
}

// ---------------------------------------------------------------------------
// Windowed flash cross-attention.  Block = (window, head), 512 thr = 16 waves.
// Each wave: one 16-query tile.  Keys streamed in 64-key LDS chunks.
// ---------------------------------------------------------------------------
#define AT_CH 64
__global__ __launch_bounds__(512)
void ow_attn_kernel(const unsigned short* __restrict__ q,
                    const unsigned short* __restrict__ k,
                    const unsigned short* __restrict__ v,
                    const unsigned short* __restrict__ vh,
                    const float* __restrict__ bias,
                    unsigned short* __restrict__ o1,
                    unsigned short* __restrict__ o2) {
    const int win = blockIdx.x;          // 0..63
    const int head = blockIdx.y;         // 0..5
    const int wy = win >> 3, wx = win & 7;

    __shared__ __align__(16) unsigned short Ks [AT_CH * 32];   // [key][dim]
    __shared__ __align__(16) unsigned short Vt [32 * AT_CH];   // [dim][key]
    __shared__ __align__(16) unsigned short VHt[32 * AT_CH];   // [dim][key]
    __shared__ __align__(16) unsigned short Ps [16][16 * 32];  // per-wave P staging

    const int tid = threadIdx.x;
    const int lane = tid & 31, wave = tid >> 5;
    const int half = lane >> 4, l16 = lane & 15;

    // query A-tile (16 queries x 32 dims) straight from global
    const int qq = wave * 16 + l16;
    const int qy = qq >> 4, qx = qq & 15;
    const size_t qbase = ((size_t)((wy * 16 + qy) * 128 + wx * 16 + qx)) * CDIM
                       + head * HD + half * 8;
    BF16x16 qa;
    qa.q[0] = *(const uint4*)(q + qbase);
    qa.q[1] = *(const uint4*)(q + qbase + 16);

    v8f o1a[2] = {}, o2a[2] = {};
    float mrow[8], lrow[8];
#pragma unroll
    for (int r = 0; r < 8; ++r) { mrow[r] = -1e30f; lrow[r] = 0.0f; }

    const float* biasH = bias + ((size_t)head * NQ + wave * 16) * NK;
    unsigned short* pw = &Ps[wave][0];

    for (int c = 0; c < NK / AT_CH; ++c) {          // 9 chunks of 64 keys
        __syncthreads();
        // cooperative gather of 64 keys (unfold w/ zero pad), V/VH transposed
        if (tid < AT_CH * 4) {
            int key = tid >> 2, ds = (tid & 3) * 8;
            int gk = c * AT_CH + key;
            int ky = gk / 24, kx = gk - ky * 24;
            int py = wy * 16 + ky - 4, px = wx * 16 + kx - 4;
            uint4 kd = {0, 0, 0, 0}, vd = {0, 0, 0, 0}, vhd = {0, 0, 0, 0};
            if (py >= 0 && py < 128 && px >= 0 && px < 128) {
                size_t base = ((size_t)(py * 128 + px)) * CDIM + head * HD + ds;
                kd  = *(const uint4*)(k  + base);
                vd  = *(const uint4*)(v  + base);
                vhd = *(const uint4*)(vh + base);
            }
            *(uint4*)(Ks + key * 32 + ds) = kd;
            const unsigned short* vp  = (const unsigned short*)&vd;
            const unsigned short* vhp = (const unsigned short*)&vhd;
#pragma unroll
            for (int j = 0; j < 8; ++j) {
                Vt [(ds + j) * AT_CH + key] = vp[j];
                VHt[(ds + j) * AT_CH + key] = vhp[j];
            }
        }
        __syncthreads();

        for (int cc = 0; cc < 2; ++cc) {            // two 32-key sub-chunks
            const int lk = cc * 32;
            v8f s[2];
#pragma unroll
            for (int st = 0; st < 2; ++st) {
                int keyn = lk + st * 16 + l16;
                BF16x16 kb;   // B operand: lane=key column, K(dim)=half*8+{0..7,16..23}
                kb.q[0] = *(const uint4*)(Ks + keyn * 32 + half * 8);
                kb.q[1] = *(const uint4*)(Ks + keyn * 32 + half * 8 + 16);
                v8f z = {};
                s[st] = WMMA_BF16(qa.v, kb.v, z);
                int gkb = c * AT_CH + lk + st * 16 + l16;
#pragma unroll
                for (int r = 0; r < 8; ++r)
                    s[st][r] += biasH[(size_t)(r + half * 8) * NK + gkb];
            }
            // online softmax update (row = r + 8*half, reduce across 16 lanes)
#pragma unroll
            for (int r = 0; r < 8; ++r) {
                float mx = fmaxf(s[0][r], s[1][r]);
                mx = fmaxf(mx, __shfl_xor(mx, 1));
                mx = fmaxf(mx, __shfl_xor(mx, 2));
                mx = fmaxf(mx, __shfl_xor(mx, 4));
                mx = fmaxf(mx, __shfl_xor(mx, 8));
                float newm = fmaxf(mrow[r], mx);
                float p0 = __expf(s[0][r] - newm);
                float p1 = __expf(s[1][r] - newm);
                float rs = p0 + p1;
                rs += __shfl_xor(rs, 1);
                rs += __shfl_xor(rs, 2);
                rs += __shfl_xor(rs, 4);
                rs += __shfl_xor(rs, 8);
                float corr = __expf(mrow[r] - newm);
                lrow[r] = lrow[r] * corr + rs;
                mrow[r] = newm;
                o1a[0][r] *= corr; o1a[1][r] *= corr;
                o2a[0][r] *= corr; o2a[1][r] *= corr;
                int mloc = r + half * 8;
                pw[mloc * 32 + l16]      = f2bf_u(p0);
                pw[mloc * 32 + 16 + l16] = f2bf_u(p1);
            }
            // compiler barrier + hardware wait: P staging is same-wave LDS
            // RAW through differently-typed accesses (TBAA can't see it)
            asm volatile("s_wait_dscnt 0" ::: "memory");
            // P as A operand (row=l16, K = half*8 + {0..7,16..23})
            BF16x16 pa;
            pa.q[0] = *(const uint4*)(pw + l16 * 32 + half * 8);
            pa.q[1] = *(const uint4*)(pw + l16 * 32 + half * 8 + 16);
#pragma unroll
            for (int j = 0; j < 2; ++j) {
                int n = j * 16 + l16;
                BF16x16 vb, vhb;   // B: lane=dim column, K(key)=half*8+{0..7,16..23}
                vb.q[0]  = *(const uint4*)(Vt  + (size_t)n * AT_CH + lk + half * 8);
                vb.q[1]  = *(const uint4*)(Vt  + (size_t)n * AT_CH + lk + half * 8 + 16);
                vhb.q[0] = *(const uint4*)(VHt + (size_t)n * AT_CH + lk + half * 8);
                vhb.q[1] = *(const uint4*)(VHt + (size_t)n * AT_CH + lk + half * 8 + 16);
                o1a[j] = WMMA_BF16(pa.v, vb.v,  o1a[j]);
                o2a[j] = WMMA_BF16(pa.v, vhb.v, o2a[j]);
            }
        }
    }

    // finalize + scatter to image layout (window_reverse folded in)
#pragma unroll
    for (int r = 0; r < 8; ++r) {
        float inv = 1.0f / lrow[r];
        int qq2 = wave * 16 + r + half * 8;
        int qy2 = qq2 >> 4, qx2 = qq2 & 15;
        size_t ob = ((size_t)((wy * 16 + qy2) * 128 + wx * 16 + qx2)) * CDIM + head * HD;
#pragma unroll
        for (int j = 0; j < 2; ++j) {
            o1[ob + j * 16 + l16] = f2bf_u(o1a[j][r] * inv);
            o2[ob + j * 16 + l16] = f2bf_u(o2a[j][r] * inv);
        }
    }
}

// ---------------------------------------------------------------------------
// MoE gating: logits + hash-gumbel softmax, top-1, accumulate mean gates
// ---------------------------------------------------------------------------
__device__ __forceinline__ float ow_u01(uint32_t x) {
    x ^= x >> 16; x *= 0x7feb352dU;
    x ^= x >> 15; x *= 0x846ca68bU;
    x ^= x >> 16;
    return (float)(x >> 8) * (1.0f / 16777216.0f) + 2.98e-8f;
}

__global__ void ow_gate_kernel(const unsigned short* __restrict__ xm,
                               const float* __restrict__ gw,
                               const float* __restrict__ gb,
                               float* __restrict__ vals,
                               int* __restrict__ idxs,
                               float* __restrict__ gatesum,
                               uint32_t seed) {
    int t = blockIdx.x * blockDim.x + threadIdx.x;
    if (t >= NTOK) return;
    float logit[NEXP];
#pragma unroll
    for (int e = 0; e < NEXP; ++e) logit[e] = gb[e];
    for (int c = 0; c < CDIM; ++c) {
        float xv = bf2f_u(xm[(size_t)t * CDIM + c]);
#pragma unroll
        for (int e = 0; e < NEXP; ++e) logit[e] += xv * gw[c * NEXP + e];
    }
#pragma unroll
    for (int e = 0; e < NEXP; ++e) {
        uint32_t h = (uint32_t)t * 0x9E3779B9u + (uint32_t)e * 0x85EBCA6Bu + seed * 0xC2B2AE35u;
        float u = ow_u01(h);
        logit[e] += -logf(-logf(u));        // gumbel
    }
    float mx = logit[0];
#pragma unroll
    for (int e = 1; e < NEXP; ++e) mx = fmaxf(mx, logit[e]);
    float ex[NEXP], sum = 0.0f;
#pragma unroll
    for (int e = 0; e < NEXP; ++e) { ex[e] = __expf(logit[e] - mx); sum += ex[e]; }
    float inv = 1.0f / sum;
    int   beste = 0;
    float bestv = ex[0] * inv;
#pragma unroll
    for (int e = 0; e < NEXP; ++e) {
        float g = ex[e] * inv;
        if (g > bestv) { bestv = g; beste = e; }
        atomicAdd(&gatesum[e], g);
    }
    vals[t] = bestv;
    idxs[t] = beste;
}

__global__ void ow_lb_kernel(const float* __restrict__ gatesum,
                             float* __restrict__ lb_acc, int slot) {
    float lb = 0.0f;
    for (int e = 0; e < NEXP; ++e) {
        float mg = gatesum[e] * (1.0f / (float)NTOK);
        lb += mg * logf(mg + 1e-8f);
    }
    lb_acc[slot] = lb;
}

__global__ void ow_finalize_lb(const float* __restrict__ lb_acc,
                               float* __restrict__ out_scalar) {
    out_scalar[0] = 0.5f * (lb_acc[0] + lb_acc[1]);
}

// combine: xr += vals[t] * outs[idx[t]][t][c]
__global__ void ow_combine_kernel(float* __restrict__ xr,
                                  const unsigned short* __restrict__ outs,
                                  const float* __restrict__ vals,
                                  const int* __restrict__ idxs) {
    int i = blockIdx.x * blockDim.x + threadIdx.x;
    const int total = NTOK * CDIM;
    if (i >= total) return;
    int t = i / CDIM;
    int e = idxs[t];
    xr[i] += vals[t] * bf2f_u(outs[(size_t)e * total + i]);
}

// ---------------------------------------------------------------------------
// Host-side orchestration
// ---------------------------------------------------------------------------
static inline void launch_gemm(hipStream_t s, const unsigned short* A, const float* B,
                               const float* bias, int M, int N, int K, int mode, float scale0,
                               unsigned short* ob0, unsigned short* ob1,
                               const float* res, float* of) {
    dim3 grid(M / 256, N / 64);
    ow_gemm_wmma<<<grid, 256, 0, s>>>(A, B, bias, M, N, K, mode, scale0, ob0, ob1, res, of);
}

extern "C" void kernel_launch(void* const* d_in, const int* in_sizes, int n_in,
                              void* d_out, int out_size, void* d_ws, size_t ws_size,
                              hipStream_t stream) {
    const float* x1   = (const float*)d_in[0];
    const float* x2   = (const float*)d_in[1];
    const int*   rpi  = (const int*)  d_in[2];
    const float* n1g  = (const float*)d_in[3];
    const float* n1b  = (const float*)d_in[4];
    const float* n2g  = (const float*)d_in[5];
    const float* n2b  = (const float*)d_in[6];
    const float* n3g  = (const float*)d_in[7];
    const float* n3b  = (const float*)d_in[8];
    const float* n4g  = (const float*)d_in[9];
    const float* n4b  = (const float*)d_in[10];
    const float* qvw  = (const float*)d_in[11];
    const float* qvb  = (const float*)d_in[12];
    const float* kvw  = (const float*)d_in[13];
    const float* kvb  = (const float*)d_in[14];
    const float* rpb  = (const float*)d_in[15];
    const float* p1w  = (const float*)d_in[16];
    const float* p1b  = (const float*)d_in[17];
    const float* p2w  = (const float*)d_in[18];
    const float* p2b  = (const float*)d_in[19];
    const float* g1w  = (const float*)d_in[20];
    const float* g1b  = (const float*)d_in[21];
    const float* m1w1 = (const float*)d_in[22];
    const float* m1b1 = (const float*)d_in[23];
    const float* m1w2 = (const float*)d_in[24];
    const float* m1b2 = (const float*)d_in[25];
    const float* g2w  = (const float*)d_in[26];
    const float* g2b  = (const float*)d_in[27];
    const float* m2w1 = (const float*)d_in[28];
    const float* m2b1 = (const float*)d_in[29];
    const float* m2w2 = (const float*)d_in[30];
    const float* m2b2 = (const float*)d_in[31];
    (void)in_sizes; (void)n_in; (void)out_size; (void)ws_size;

    // workspace carve-up
    char* W = (char*)d_ws;
    size_t off = 0;
    auto carve = [&](size_t bytes) -> char* {
        char* p = W + off;
        off += (bytes + 255) & ~(size_t)255;
        return p;
    };
    const size_t nc_bf = (size_t)NTOK * CDIM * 2;
    unsigned short* q_bf   = (unsigned short*)carve(nc_bf);
    unsigned short* k_bf   = (unsigned short*)carve(nc_bf);
    unsigned short* v_bf   = (unsigned short*)carve(nc_bf);
    unsigned short* vh_bf  = (unsigned short*)carve(nc_bf);
    unsigned short* x1n_bf = (unsigned short*)carve(nc_bf);
    unsigned short* x2n_bf = (unsigned short*)carve(nc_bf);
    float*          biasP  = (float*)carve((size_t)NHEAD * NQ * NK * 4);
    unsigned short* o1_bf  = (unsigned short*)carve(nc_bf);
    unsigned short* o2_bf  = (unsigned short*)carve(nc_bf);
    unsigned short* xm_bf  = (unsigned short*)carve(nc_bf);
    unsigned short* hid_bf = (unsigned short*)carve((size_t)NTOK * MH_DIM * 2);
    unsigned short* outsbf = (unsigned short*)carve((size_t)NEXP * NTOK * CDIM * 2);
    float*          vals   = (float*)carve((size_t)NTOK * 4);
    int*            idxs   = (int*)  carve((size_t)NTOK * 4);
    float*          gatesum= (float*)carve(256);
    float*          lb_acc = (float*)carve(256);

    float* x1r = (float*)d_out;                          // N*C
    float* x2r = (float*)d_out + (size_t)NTOK * CDIM;    // N*C
    float* lbo = (float*)d_out + (size_t)2 * NTOK * CDIM;

    // 1) LayerNorms -> bf16
    ow_ln_kernel<<<NTOK, CDIM, 0, stream>>>(x1, n1g, n1b, x1n_bf);
    ow_ln_kernel<<<NTOK, CDIM, 0, stream>>>(x2, n2g, n2b, x2n_bf);

    // 2) QV / KV projections (WMMA), split + scale q
    launch_gemm(stream, x1n_bf, qvw, qvb, NTOK, 2 * CDIM, CDIM, 0, SCALE_Q, q_bf, vh_bf, nullptr, nullptr);
    launch_gemm(stream, x2n_bf, kvw, kvb, NTOK, 2 * CDIM, CDIM, 0, 1.0f,    k_bf, v_bf,  nullptr, nullptr);

    // 3) attention bias table
    {
        int total = NHEAD * NQ * NK;
        ow_bias_pre<<<(total + 255) / 256, 256, 0, stream>>>(rpi, rpb, biasP);
    }

    // 4) windowed flash cross-attention (WMMA)
    ow_attn_kernel<<<dim3(NWIN, NHEAD), 512, 0, stream>>>(q_bf, k_bf, v_bf, vh_bf, biasP, o1_bf, o2_bf);

    // 5) output projections + residual -> x1r/x2r (f32, in d_out)
    launch_gemm(stream, o1_bf, p1w, p1b, NTOK, CDIM, CDIM, 1, 1.0f, nullptr, nullptr, x1, x1r);
    launch_gemm(stream, o2_bf, p2w, p2b, NTOK, CDIM, CDIM, 1, 1.0f, nullptr, nullptr, x2, x2r);

    // 6) MoE blocks
    for (int moe = 0; moe < 2; ++moe) {
        float* xr = moe ? x2r : x1r;
        const float* ng = moe ? n4g : n3g;
        const float* nb = moe ? n4b : n3b;
        const float* gw = moe ? g2w : g1w;
        const float* gb = moe ? g2b : g1b;
        const float* w1 = moe ? m2w1 : m1w1;
        const float* b1 = moe ? m2b1 : m1b1;
        const float* w2 = moe ? m2w2 : m1w2;
        const float* b2 = moe ? m2b2 : m1b2;
        uint32_t seed = moe ? 102u : 101u;

        ow_ln_kernel<<<NTOK, CDIM, 0, stream>>>(xr, ng, nb, xm_bf);
        (void)hipMemsetAsync(gatesum, 0, NEXP * sizeof(float), stream);
        ow_gate_kernel<<<NTOK / 256, 256, 0, stream>>>(xm_bf, gw, gb, vals, idxs, gatesum, seed);
        ow_lb_kernel<<<1, 1, 0, stream>>>(gatesum, lb_acc, moe);

        for (int e = 0; e < NEXP; ++e) {
            // hid = gelu(xm @ w1[e] + b1[e])   (16384x192x384)
            launch_gemm(stream, xm_bf, w1 + (size_t)e * CDIM * MH_DIM, b1 + (size_t)e * MH_DIM,
                        NTOK, MH_DIM, CDIM, 2, 1.0f, hid_bf, nullptr, nullptr, nullptr);
            // outs[e] = hid @ w2[e] + b2[e]    (16384x384x192)
            launch_gemm(stream, hid_bf, w2 + (size_t)e * MH_DIM * CDIM, b2 + (size_t)e * CDIM,
                        NTOK, CDIM, MH_DIM, 3, 1.0f,
                        outsbf + (size_t)e * NTOK * CDIM, nullptr, nullptr, nullptr);
        }
        ow_combine_kernel<<<(NTOK * CDIM + 255) / 256, 256, 0, stream>>>(xr, outsbf, vals, idxs);
    }

    // 7) load-balance scalar
    ow_finalize_lb<<<1, 1, 0, stream>>>(lb_acc, lbo);
}
